// GroupedQueryAttention_3539053052486
// MI455X (gfx1250) — matrix operand order repaired and checked by reading, not compiled, not run on hardware
//
#include <hip/hip_runtime.h>

typedef __attribute__((ext_vector_type(16))) __bf16 bf16x16;
typedef __attribute__((ext_vector_type(8)))  __bf16 bf16x8;
typedef __attribute__((ext_vector_type(8)))  float  f32x8;

#define S_LEN 4096
#define D_IN  2048
#define NHEAD 16
#define NGRP  4
#define DHEAD 128
#define QKV_N 3072   // 2048 q + 512 k + 512 v

// ---------- WMMA fragment helpers (CDNA5 16x16x32 bf16 layouts, ISA 7.12.2) ----------

// A fragment: M=16 rows (lane&15), K=32. elems 0..7 <- K = 8*half.. ; elems 8..15 <- K = 16+8*half..
static __device__ __forceinline__ bf16x16 load_frag_a(const __bf16* base, int ld, int lane) {
  const int row  = lane & 15;
  const int half = (lane >> 4) & 1;
  const __bf16* p = base + (size_t)row * ld + half * 8;
  bf16x8 lo = *(const bf16x8*)(p);
  bf16x8 hi = *(const bf16x8*)(p + 16);
  bf16x16 f;
#pragma unroll
  for (int i = 0; i < 8; ++i) { f[i] = lo[i]; f[i + 8] = hi[i]; }
  return f;
}

// B fragment: K=32, N=16 cols (lane&15). elems 0..7 <- K = 8*half.. ; elems 8..15 <- K = 16+8*half..
// Source is row-major [N][K] (i.e. W[n][k] for C = A @ W^T).
static __device__ __forceinline__ bf16x16 load_frag_b(const __bf16* base, int ld, int lane) {
  const int n    = lane & 15;
  const int half = (lane >> 4) & 1;
  const __bf16* p = base + (size_t)n * ld + half * 8;
  bf16x8 lo = *(const bf16x8*)(p);
  bf16x8 hi = *(const bf16x8*)(p + 16);
  bf16x16 f;
#pragma unroll
  for (int i = 0; i < 8; ++i) { f[i] = lo[i]; f[i + 8] = hi[i]; }
  return f;
}

static __device__ __forceinline__ f32x8 wmma_bf16(bf16x16 a, bf16x16 b, f32x8 c) {
  return __builtin_amdgcn_wmma_f32_16x16x32_bf16(false, a, false, b, (short)0, c, false, false);
}

// ---------- elementwise converts ----------

__global__ void k_f32_to_bf16(const float* __restrict__ src, __bf16* __restrict__ dst, int n) {
  int i = blockIdx.x * blockDim.x + threadIdx.x;
  if (i < n) dst[i] = (__bf16)src[i];
}

// strided slice convert: dst[s*ncol + c] = src[s*srcld + colbase + c]
__global__ void k_slice_to_bf16(const float* __restrict__ src, __bf16* __restrict__ dst,
                                int ncol, int srcld, int colbase) {
  int i = blockIdx.x * blockDim.x + threadIdx.x;
  if (i >= S_LEN * ncol) return;
  int s = i / ncol, c = i - s * ncol;
  dst[(size_t)s * ncol + c] = (__bf16)src[(size_t)s * srcld + colbase + c];
}

// RoPE + scale + bf16 convert. src = qkv fp32 [S][3072], cols [colbase, colbase+ncol)
__global__ void k_rope_bf16(const float* __restrict__ qkv, const float* __restrict__ cosT,
                            const float* __restrict__ sinT, __bf16* __restrict__ dst,
                            int ncol, int colbase, float scale) {
  int i = blockIdx.x * blockDim.x + threadIdx.x;
  if (i >= S_LEN * ncol) return;
  int s = i / ncol, c = i - s * ncol;
  int d = c & (DHEAD - 1);
  const float* row = qkv + (size_t)s * QKV_N + colbase;
  float v   = row[c];
  float rot = (d < DHEAD / 2) ? -row[c + DHEAD / 2] : row[c - DHEAD / 2];
  float cv = cosT[s * DHEAD + d], sv = sinT[s * DHEAD + d];
  dst[(size_t)s * ncol + c] = (__bf16)((v * cv + rot * sv) * scale);
}

// ---------- GEMM: C[M][N] (f32) = A[M][K] (bf16) @ B[N][K]^T (bf16) ----------
// block = 128 thr = 4 waves (2x2), block tile 128x128, wave tile 64x64.
// 16 WMMAs per 16 b128 loads per K-step: ~512B of L2 traffic per WMMA (32 FLOP/B).
__global__ __launch_bounds__(128) void k_gemm_bf16nt(const __bf16* __restrict__ A,
                                                     const __bf16* __restrict__ B,
                                                     float* __restrict__ C,
                                                     int M, int N, int K) {
  const int lane = threadIdx.x & 31;
  const int wave = threadIdx.x >> 5;                 // 0..3
  const int m0 = blockIdx.y * 128 + (wave >> 1) * 64;
  const int n0 = blockIdx.x * 128 + (wave & 1) * 64;
  f32x8 acc[4][4] = {};
  const __bf16* Arow[4];
#pragma unroll
  for (int i = 0; i < 4; ++i) Arow[i] = A + (size_t)(m0 + 16 * i) * K;
  for (int k = 0; k < K; k += 32) {
    __builtin_prefetch(Arow[0] + k + 256, 0, 1);     // global_prefetch_b8
    bf16x16 a[4];
#pragma unroll
    for (int i = 0; i < 4; ++i) a[i] = load_frag_a(Arow[i] + k, K, lane);
#pragma unroll
    for (int j = 0; j < 4; ++j) {
      bf16x16 bj = load_frag_b(B + (size_t)(n0 + 16 * j) * K + k, K, lane);
#pragma unroll
      for (int i = 0; i < 4; ++i) acc[i][j] = wmma_bf16(a[i], bj, acc[i][j]);
    }
  }
  const int half = lane >> 4, col = lane & 15;
#pragma unroll
  for (int i = 0; i < 4; ++i)
#pragma unroll
    for (int j = 0; j < 4; ++j)
#pragma unroll
      for (int e = 0; e < 8; ++e)
        C[(size_t)(m0 + 16 * i + e + 8 * half) * N + (n0 + 16 * j + col)] = acc[i][j][e];
}

// ---------- Flash attention: grid (S/64, H), 128 thr = 4 waves, wave = 16 q rows ----------
__global__ __launch_bounds__(128) void k_attn(const __bf16* __restrict__ qb,
                                              const __bf16* __restrict__ kb,
                                              const __bf16* __restrict__ vb,
                                              __bf16* __restrict__ ctxb) {
  __shared__ __bf16 ldsK[32 * DHEAD];      // K tile,   [kv=32][d=128] row-major
  __shared__ __bf16 ldsVt[DHEAD * 32];     // V tile^T, [d=128][kv=32] row-major
  __shared__ __bf16 ldsP[4][16 * 32];      // per-wave P scratch [16 q][32 kv]

  const int h    = blockIdx.y;
  const int qblk = blockIdx.x;
  const int g    = h >> 2;                 // GQA group (H/G = 4)
  const int lane = threadIdx.x & 31;
  const int wave = threadIdx.x >> 5;
  const int qrow0 = qblk * 64 + wave * 16;
  const int half = lane >> 4, ln = lane & 15;

  // Q fragments for this wave (16 rows x 128 d = 4 K-slabs), 1/sqrt(D) pre-folded.
  bf16x16 qf[4];
  const __bf16* qbase = qb + (size_t)qrow0 * (NHEAD * DHEAD) + h * DHEAD;
#pragma unroll
  for (int t = 0; t < 4; ++t) qf[t] = load_frag_a(qbase + t * 32, NHEAD * DHEAD, lane);

  f32x8 ctx[8] = {};
  float mrow[8], lrow[8];
#pragma unroll
  for (int e = 0; e < 8; ++e) { mrow[e] = -1e30f; lrow[e] = 0.0f; }

  const int nkv = qblk * 64 + 64;          // causal upper bound (exclusive)
  for (int kv0 = 0; kv0 < nkv; kv0 += 32) {
    __syncthreads();                       // previous-iteration readers done
    // stage K tile [32][128]
    for (int c = threadIdx.x; c < 512; c += 128) {
      int r = c >> 4, cc = (c & 15) * 8;
      *(bf16x8*)(ldsK + r * DHEAD + cc) =
          *(const bf16x8*)(kb + (size_t)(kv0 + r) * (NGRP * DHEAD) + g * DHEAD + cc);
    }
    // stage V tile transposed -> [128][32]
    for (int c = threadIdx.x; c < 512; c += 128) {
      int r = c >> 4, cc = (c & 15) * 8;
      bf16x8 vv = *(const bf16x8*)(vb + (size_t)(kv0 + r) * (NGRP * DHEAD) + g * DHEAD + cc);
#pragma unroll
      for (int e = 0; e < 8; ++e) ldsVt[(cc + e) * 32 + r] = vv[e];
    }
    __syncthreads();

    // scores S = Q @ K^T : 16 x 32 (two 16x16 accumulators)
    f32x8 s0 = {}, s1 = {};
#pragma unroll
    for (int t = 0; t < 4; ++t) {
      bf16x16 b0 = load_frag_b(ldsK + t * 32, DHEAD, lane);              // kv 0..15
      bf16x16 b1 = load_frag_b(ldsK + 16 * DHEAD + t * 32, DHEAD, lane); // kv 16..31
      s0 = wmma_bf16(qf[t], b0, s0);
      s1 = wmma_bf16(qf[t], b1, s1);
    }

    // online softmax (rows live in C-layout: elem e <-> row e+8*half, lane<->col)
#pragma unroll
    for (int e = 0; e < 8; ++e) {
      const int mg = qrow0 + e + 8 * half;
      const int c0 = kv0 + ln, c1 = kv0 + 16 + ln;
      float v0 = (c0 > mg) ? -1e30f : s0[e];
      float v1 = (c1 > mg) ? -1e30f : s1[e];
      float mx = fmaxf(v0, v1);
#pragma unroll
      for (int off = 1; off < 16; off <<= 1) mx = fmaxf(mx, __shfl_xor(mx, off, 32));
      float mnew  = fmaxf(mrow[e], mx);
      float alpha = __expf(mrow[e] - mnew);
      float p0 = __expf(v0 - mnew);
      float p1 = __expf(v1 - mnew);
      float ps = p0 + p1;
#pragma unroll
      for (int off = 1; off < 16; off <<= 1) ps += __shfl_xor(ps, off, 32);
      lrow[e] = lrow[e] * alpha + ps;
      mrow[e] = mnew;
#pragma unroll
      for (int j = 0; j < 8; ++j) ctx[j][e] *= alpha;
      // P -> bf16 into wave-private LDS scratch (A-layout source, row-major [16][32])
      ldsP[wave][(e + 8 * half) * 32 + ln]      = (__bf16)p0;
      ldsP[wave][(e + 8 * half) * 32 + 16 + ln] = (__bf16)p1;
    }

    // ctx += P(16x32) @ V(32x128)
    bf16x16 pf = load_frag_a(&ldsP[wave][0], 32, lane);
#pragma unroll
    for (int j = 0; j < 8; ++j) {
      bf16x16 bv = load_frag_b(ldsVt + (16 * j) * 32, 32, lane);
      ctx[j] = wmma_bf16(pf, bv, ctx[j]);
    }
  }

  // normalize and emit bf16 ctx at [s][h*128 + d]
#pragma unroll
  for (int e = 0; e < 8; ++e) {
    float rinv = 1.0f / lrow[e];
    const size_t rowoff = (size_t)(qrow0 + e + 8 * half) * (NHEAD * DHEAD) + h * DHEAD;
#pragma unroll
    for (int j = 0; j < 8; ++j)
      ctxb[rowoff + 16 * j + ln] = (__bf16)(ctx[j][e] * rinv);
  }
}

// ---------- host ----------

static inline size_t align256(size_t x) { return (x + 255) & ~(size_t)255; }

extern "C" void kernel_launch(void* const* d_in, const int* in_sizes, int n_in,
                              void* d_out, int out_size, void* d_ws, size_t ws_size,
                              hipStream_t stream) {
  const float* x    = (const float*)d_in[0];
  // d_in[1] = mask (bool) — causal mask computed analytically, unused
  const float* cosT = (const float*)d_in[2];
  const float* sinT = (const float*)d_in[3];
  const float* Wq   = (const float*)d_in[4];
  const float* Wk   = (const float*)d_in[5];
  const float* Wv   = (const float*)d_in[6];
  const float* Wo   = (const float*)d_in[7];
  float* out = (float*)d_out;

  char* ws = (char*)d_ws;
  size_t off = 0;
  __bf16* xb     = (__bf16*)(ws + off); off = align256(off + (size_t)S_LEN * D_IN * 2);
  __bf16* wqkv_b = (__bf16*)(ws + off); off = align256(off + (size_t)QKV_N * D_IN * 2);
  __bf16* wo_b   = (__bf16*)(ws + off); off = align256(off + (size_t)D_IN * D_IN * 2);
  float*  qkv_f  = (float*) (ws + off); off = align256(off + (size_t)S_LEN * QKV_N * 4);
  __bf16* qb     = (__bf16*)(ws + off); off = align256(off + (size_t)S_LEN * NHEAD * DHEAD * 2);
  __bf16* kb     = (__bf16*)(ws + off); off = align256(off + (size_t)S_LEN * NGRP * DHEAD * 2);
  __bf16* vb     = (__bf16*)(ws + off); off = align256(off + (size_t)S_LEN * NGRP * DHEAD * 2);
  __bf16* ctxb   = (__bf16*)(ws + off); off = align256(off + (size_t)S_LEN * NHEAD * DHEAD * 2);
  (void)ws_size; (void)in_sizes; (void)n_in; (void)out_size;

  const int T = 256;
  auto blocks = [&](long n) { return (unsigned)((n + T - 1) / T); };

  // 1) bf16 converts (Wq/Wk/Wv packed row-contiguous into one [3072][2048] weight)
  long nx = (long)S_LEN * D_IN;
  k_f32_to_bf16<<<blocks(nx), T, 0, stream>>>(x, xb, (int)nx);
  long nwq = (long)D_IN * D_IN, nwk = (long)(NGRP * DHEAD) * D_IN;
  k_f32_to_bf16<<<blocks(nwq), T, 0, stream>>>(Wq, wqkv_b, (int)nwq);
  k_f32_to_bf16<<<blocks(nwk), T, 0, stream>>>(Wk, wqkv_b + nwq, (int)nwk);
  k_f32_to_bf16<<<blocks(nwk), T, 0, stream>>>(Wv, wqkv_b + nwq + nwk, (int)nwk);
  k_f32_to_bf16<<<blocks(nwq), T, 0, stream>>>(Wo, wo_b, (int)nwq);

  // 2) fused QKV GEMM: [4096][3072] = xb @ wqkv_b^T
  {
    dim3 grid(QKV_N / 128, S_LEN / 128);
    k_gemm_bf16nt<<<grid, 128, 0, stream>>>(xb, wqkv_b, qkv_f, S_LEN, QKV_N, D_IN);
  }

  // 3) RoPE (+1/sqrt(D) folded into q) and V convert
  const float scale = 0.088388347648318447f;  // 1/sqrt(128)
  long nq = (long)S_LEN * NHEAD * DHEAD, nk = (long)S_LEN * NGRP * DHEAD;
  k_rope_bf16<<<blocks(nq), T, 0, stream>>>(qkv_f, cosT, sinT, qb, NHEAD * DHEAD, 0, scale);
  k_rope_bf16<<<blocks(nk), T, 0, stream>>>(qkv_f, cosT, sinT, kb, NGRP * DHEAD, D_IN, 1.0f);
  k_slice_to_bf16<<<blocks(nk), T, 0, stream>>>(qkv_f, vb, NGRP * DHEAD, QKV_N,
                                                D_IN + NGRP * DHEAD);

  // 4) flash attention
  {
    dim3 grid(S_LEN / 64, NHEAD);
    k_attn<<<grid, 128, 0, stream>>>(qb, kb, vb, ctxb);
  }

  // 5) output projection: out[4096][2048] = ctxb @ wo_b^T
  {
    dim3 grid(D_IN / 128, S_LEN / 128);
    k_gemm_bf16nt<<<grid, 128, 0, stream>>>(ctxb, wo_b, out, S_LEN, D_IN, D_IN);
  }
}
